// BlockAttention_10866267259494
// MI455X (gfx1250) — compile-verified, matrix operand order checked by reading, not run on hardware
//
#include <hip/hip_runtime.h>

typedef __attribute__((ext_vector_type(16))) __bf16 v16bf;
typedef __attribute__((ext_vector_type(8)))  __bf16 v8bf;
typedef __attribute__((ext_vector_type(8)))  float  v8f;

#define BM 128
#define BN 128
#define BK 32
#define LDSS 40   // padded stride (bf16) -> 20 dwords: conflict-free 16B fragment reads

// pack 4 f32 -> 4 bf16 in one 64-bit value (single ds_store_b64)
union bf4pack { __bf16 b[4]; uint2 u; };

// ---------------------------------------------------------------------------
// C[M,N] = A[M,K] @ B[K,N] + bias[N]   (fp32 memory, bf16 WMMA, f32 accum)
// M%128==0, N%128==0, K%32==0 at all call sites -> no bounds checks.
// Double-buffered LDS ping-pong, one barrier per K-tile.
// ---------------------------------------------------------------------------
__global__ __launch_bounds__(256)
void gemm_bf16_wmma(const float* __restrict__ A, const float* __restrict__ Bw,
                    const float* __restrict__ bias, float* __restrict__ C,
                    int M, int N, int K)
{
    __shared__ __bf16 As[2][BM][LDSS];
    __shared__ __bf16 Bs[2][BN][LDSS];

    const int tid   = threadIdx.x;
    const int lane  = tid & 31;
    const int wave  = tid >> 5;     // 0..7
    const int waveM = wave >> 2;    // 0..1  : 64 rows
    const int waveN = wave & 3;     // 0..3  : 32 cols

    const int blockN = blockIdx.x * BN;
    const int blockM = blockIdx.y * BM;

    const int m16 = lane & 15;           // row (A) / col (B,C) within 16
    const int kg8 = (lane >> 4) << 3;    // K-half per ISA 16-bit fragment layout

    // staging map: A -> thread owns 4 rows x 4 k (float4 along K)
    const int ar = tid >> 3;             // 0..31
    const int ac = (tid & 7) << 2;       // 0..28
    // staging map: B -> thread owns 1 col x 16 k (scalar loads, coalesced over n)
    const int bn    = tid & 127;         // 0..127
    const int khalf = tid >> 7;          // 0..1 -> k base 0 / 16

    v8f acc[4][2] = {};

    const int nT = K / BK;

    // ---- prologue: stage tile 0 into buffer 0 ----
    {
        #pragma unroll
        for (int rr = 0; rr < 4; ++rr) {
            const float4 v = *(const float4*)(A + (size_t)(blockM + ar + rr * 32) * K + ac);
            bf4pack p;
            p.b[0] = (__bf16)v.x; p.b[1] = (__bf16)v.y;
            p.b[2] = (__bf16)v.z; p.b[3] = (__bf16)v.w;
            *(uint2*)&As[0][ar + rr * 32][ac] = p.u;
        }
        #pragma unroll
        for (int g = 0; g < 4; ++g) {
            bf4pack p;
            #pragma unroll
            for (int i = 0; i < 4; ++i)
                p.b[i] = (__bf16)Bw[(size_t)(khalf * 16 + g * 4 + i) * N + (blockN + bn)];
            *(uint2*)&Bs[0][bn][khalf * 16 + g * 4] = p.u;
        }
    }
    __syncthreads();

    for (int kt = 0; kt < nT; ++kt) {
        const int  buf     = kt & 1;
        const bool hasNext = (kt + 1) < nT;
        const int  kn      = (kt + 1) * BK;

        // ---- issue next tile's global loads (latency hidden behind WMMAs) ----
        float4 aReg[4];
        float  bReg[16];
        if (hasNext) {
            #pragma unroll
            for (int rr = 0; rr < 4; ++rr)
                aReg[rr] = *(const float4*)(A + (size_t)(blockM + ar + rr * 32) * K + (kn + ac));
            #pragma unroll
            for (int i = 0; i < 16; ++i)
                bReg[i] = Bw[(size_t)(kn + khalf * 16 + i) * N + (blockN + bn)];
        }

        // ---- compute on current buffer ----
        v16bf afrag[4], bfrag[2];
        #pragma unroll
        for (int tm = 0; tm < 4; ++tm) {
            const int row = waveM * 64 + tm * 16 + m16;
            v8bf lo = *(const v8bf*)&As[buf][row][kg8];
            v8bf hi = *(const v8bf*)&As[buf][row][16 + kg8];
            afrag[tm] = __builtin_shufflevector(lo, hi, 0,1,2,3,4,5,6,7,8,9,10,11,12,13,14,15);
        }
        #pragma unroll
        for (int tn = 0; tn < 2; ++tn) {
            const int col = waveN * 32 + tn * 16 + m16;
            v8bf lo = *(const v8bf*)&Bs[buf][col][kg8];
            v8bf hi = *(const v8bf*)&Bs[buf][col][16 + kg8];
            bfrag[tn] = __builtin_shufflevector(lo, hi, 0,1,2,3,4,5,6,7,8,9,10,11,12,13,14,15);
        }
        #pragma unroll
        for (int tm = 0; tm < 4; ++tm)
            #pragma unroll
            for (int tn = 0; tn < 2; ++tn)
                acc[tm][tn] = __builtin_amdgcn_wmma_f32_16x16x32_bf16(
                    false, afrag[tm], false, bfrag[tn],
                    (short)0, acc[tm][tn], false, false);

        // ---- convert + store next tile into alternate buffer ----
        if (hasNext) {
            const int nbuf = buf ^ 1;
            #pragma unroll
            for (int rr = 0; rr < 4; ++rr) {
                bf4pack p;
                p.b[0] = (__bf16)aReg[rr].x; p.b[1] = (__bf16)aReg[rr].y;
                p.b[2] = (__bf16)aReg[rr].z; p.b[3] = (__bf16)aReg[rr].w;
                *(uint2*)&As[nbuf][ar + rr * 32][ac] = p.u;
            }
            #pragma unroll
            for (int g = 0; g < 4; ++g) {
                bf4pack p;
                #pragma unroll
                for (int i = 0; i < 4; ++i) p.b[i] = (__bf16)bReg[g * 4 + i];
                *(uint2*)&Bs[nbuf][bn][khalf * 16 + g * 4] = p.u;
            }
        }
        __syncthreads();
    }

    // ---- epilogue: 64-bit base hoisted once, 32-bit offsets per store ----
    float* Cp = C + (size_t)(blockM + waveM * 64 + ((lane >> 4) << 3)) * N
                  + (blockN + waveN * 32 + m16);
    #pragma unroll
    for (int tn = 0; tn < 2; ++tn) {
        const float bv = bias[blockN + waveN * 32 + tn * 16 + m16];
        #pragma unroll
        for (int tm = 0; tm < 4; ++tm) {
            #pragma unroll
            for (int r = 0; r < 8; ++r)
                Cp[(tm * 16 + r) * N + tn * 16] = acc[tm][tn][r] + bv;
        }
    }
}

// ---------------------------------------------------------------------------
// Block attention: one wave32 per (b, h, n) block; 4x64 q/k/v; each lane owns
// 2 head-dim channels; 4x4 scores via wave32 xor-butterfly reductions.
// ---------------------------------------------------------------------------
#define L_SEQ 4096
#define D_MOD 1024
#define N_H   16
#define D_H   64
#define N_BLK (L_SEQ / 4)   // 1024

__global__ __launch_bounds__(256)
void block_attn(const float* __restrict__ qkv, float* __restrict__ o)
{
    const int lane = threadIdx.x & 31;
    const int gw   = blockIdx.x * 8 + (threadIdx.x >> 5);

    const int b   = gw / (N_H * N_BLK);
    const int rem = gw % (N_H * N_BLK);
    const int h   = rem / N_BLK;
    const int nb  = rem % N_BLK;

    const size_t rowBase = ((size_t)b * L_SEQ + (size_t)nb * 4) * (3 * D_MOD) + (size_t)h * (3 * D_H);
    const int dc = lane * 2;

    float2 q[4], k[4], v[4];
    #pragma unroll
    for (int i = 0; i < 4; ++i) {
        const float* p = qkv + rowBase + (size_t)i * (3 * D_MOD);
        q[i] = *(const float2*)(p + dc);
        k[i] = *(const float2*)(p + D_H + dc);
        v[i] = *(const float2*)(p + 2 * D_H + dc);
    }

    float s[4][4];
    #pragma unroll
    for (int i = 0; i < 4; ++i)
        #pragma unroll
        for (int j = 0; j < 4; ++j) {
            float p = q[i].x * k[j].x + q[i].y * k[j].y;
            #pragma unroll
            for (int off = 16; off >= 1; off >>= 1)
                p += __shfl_xor(p, off, 32);
            s[i][j] = p * 0.125f;   // 64^{-1/2}
        }

    float pr[4][4];
    #pragma unroll
    for (int i = 0; i < 4; ++i) {
        const float m = fmaxf(fmaxf(s[i][0], s[i][1]), fmaxf(s[i][2], s[i][3]));
        float sum = 0.f;
        #pragma unroll
        for (int j = 0; j < 4; ++j) { pr[i][j] = __expf(s[i][j] - m); sum += pr[i][j]; }
        const float inv = 1.0f / sum;
        #pragma unroll
        for (int j = 0; j < 4; ++j) pr[i][j] *= inv;
    }

    #pragma unroll
    for (int i = 0; i < 4; ++i) {
        float2 acc; acc.x = 0.f; acc.y = 0.f;
        #pragma unroll
        for (int j = 0; j < 4; ++j) {
            acc.x += pr[i][j] * v[j].x;
            acc.y += pr[i][j] * v[j].y;
        }
        float* op = o + ((size_t)b * L_SEQ + (size_t)nb * 4 + i) * D_MOD + (size_t)h * D_H + dc;
        *(float2*)op = acc;
    }
}

// ---------------------------------------------------------------------------
extern "C" void kernel_launch(void* const* d_in, const int* in_sizes, int n_in,
                              void* d_out, int out_size, void* d_ws, size_t ws_size,
                              hipStream_t stream)
{
    const float* x      = (const float*)d_in[0];
    const float* W_qkv  = (const float*)d_in[1];
    const float* b_qkv  = (const float*)d_in[2];
    const float* W_proj = (const float*)d_in[3];
    const float* b_proj = (const float*)d_in[4];
    float* out = (float*)d_out;

    const int M = 8 * 4096;   // 32768
    const int D = 1024;

    float* qkv = (float*)d_ws;                                              // 32768 x 3072 f32
    float* o   = (float*)((char*)d_ws + (size_t)M * 3 * D * sizeof(float)); // 32768 x 1024 f32

    gemm_bf16_wmma<<<dim3((3 * D) / BN, M / BM), 256, 0, stream>>>(
        x, W_qkv, b_qkv, qkv, M, 3 * D, D);

    block_attn<<<(8 * N_H * N_BLK) / 8, 256, 0, stream>>>(qkv, o);

    gemm_bf16_wmma<<<dim3(D / BN, M / BM), 256, 0, stream>>>(
        o, W_proj, b_proj, out, M, D, D);
}
